// CausalSelfAttention_62182536511589
// MI455X (gfx1250) — compile-verified, matrix operand order checked by reading, not run on hardware
//
#include <hip/hip_runtime.h>

// ---------------------------------------------------------------------------
// Causal self-attention for MI455X (gfx1250, wave32, WMMA).
// fp32 in/out; internal compute in f16 with fp32 WMMA accumulation.
// ---------------------------------------------------------------------------

typedef __attribute__((ext_vector_type(16))) _Float16 v16h;
typedef __attribute__((ext_vector_type(8)))  float    v8f;
typedef int i32x4 __attribute__((vector_size(16)));

#define D_MODEL 1024
#define NHEADS  16
#define HDIM    64
#define SEQ     2048
#define BATCH   4
#define MTOT    (BATCH * SEQ)   // 8192 rows

// Load a 16x32 f16 fragment (A-layout or symmetric B-layout) where the WMMA
// K-dimension is contiguous in memory at `p` (p = row/col start + k-base).
// Per CDNA5 ISA 16-bit A layout: lane half = lane>>4,
//   VGPR v holds K = (v/4)*16 + half*8 + (v%4)*2  (+1 for hi half of dword).
// Per half this is two contiguous 16B chunks -> compiler emits 2x b128 loads.
__device__ __forceinline__ v16h frag_load(const _Float16* p, int half) {
  union { v16h h; unsigned int u[8]; } r;
#pragma unroll
  for (int v = 0; v < 8; ++v) {
    int k = ((v >> 2) << 4) + (half << 3) + ((v & 3) << 1);
    r.u[v] = *(const unsigned int*)(p + k);
  }
  return r.h;
}

// Async copy 16B global -> LDS (CDNA5 GLOBAL_LOAD_ASYNC_TO_LDS_B128 when the
// toolchain exposes the builtin; plain copy fallback keeps compilation safe).
__device__ __forceinline__ void async_copy_b128(const _Float16* g, _Float16* l) {
#if __has_builtin(__builtin_amdgcn_global_load_async_to_lds_b128)
  __builtin_amdgcn_global_load_async_to_lds_b128(
      (__attribute__((address_space(1))) i32x4*)g,
      (__attribute__((address_space(3))) i32x4*)l, 0, 0);
#else
  *(uint4*)l = *(const uint4*)g;
#endif
}

__device__ __forceinline__ void wait_async0() {
#if __has_builtin(__builtin_amdgcn_s_wait_asynccnt)
  __builtin_amdgcn_s_wait_asynccnt(0);
#else
  asm volatile("s_wait_asynccnt 0" ::: "memory");
#endif
}

__global__ void f32_to_f16(const float* __restrict__ s, _Float16* __restrict__ d, int n) {
  int i = blockIdx.x * blockDim.x + threadIdx.x;
  int stride = gridDim.x * blockDim.x;
  for (; i < n; i += stride) d[i] = (_Float16)s[i];
}

// GEMM: out[m,n] = sum_k A[m,k] * W[n,k] + bias[n]   (nn.Linear semantics)
// Block = 4 waves computing a 32x256 tile; each wave owns 32x64 (8 WMMA acc).
// The 32x1024 A tile (contiguous 64KB) is staged once into LDS with async
// loads and shared by all waves; B (weights) streams from L2.
// mode 0: store Q (bh,s,hd) scaled by 1/sqrt(HDIM)
// mode 1: store K (bh,s,hd)
// mode 2: store V transposed (bh,hd,s)
// mode 3: store fp32 row-major to outF (final projection)
__global__ __launch_bounds__(128)
void gemm32(const _Float16* __restrict__ A, const _Float16* __restrict__ W,
            const float* __restrict__ bias, _Float16* __restrict__ outH,
            float* __restrict__ outF, int mode) {
  __shared__ __align__(16) _Float16 Alds[32 * D_MODEL];   // 64 KB
  const int tid  = threadIdx.x;
  const int lane = tid & 31;
  const int wave = tid >> 5;
  const int half = lane >> 4;
  const int ln   = lane & 15;
  const int m0   = blockIdx.x * 32;
  const int n0   = blockIdx.y * 256 + wave * 64;

  // Stage the whole 32x1024 A tile (64KB, contiguous rows) into LDS.
  {
    const _Float16* Ablk = A + (size_t)m0 * D_MODEL;
#pragma unroll 4
    for (int it = 0; it < 32; ++it) {
      int idx = (it * 128 + tid) * 8;             // 8 halfs = 16B per lane
      async_copy_b128(Ablk + idx, &Alds[idx]);
    }
    wait_async0();
    __syncthreads();
  }

  v8f acc[2][4] = {};
  const _Float16* aL0 = &Alds[(size_t)ln * D_MODEL];
  const _Float16* aL1 = &Alds[(size_t)(16 + ln) * D_MODEL];
  const _Float16* b0p = W + (size_t)(n0 + ln) * D_MODEL;
  const _Float16* b1p = W + (size_t)(n0 + 16 + ln) * D_MODEL;
  const _Float16* b2p = W + (size_t)(n0 + 32 + ln) * D_MODEL;
  const _Float16* b3p = W + (size_t)(n0 + 48 + ln) * D_MODEL;

  for (int k0 = 0; k0 < D_MODEL; k0 += 32) {
    v16h a0 = frag_load(aL0 + k0, half);          // ds_load from LDS
    v16h a1 = frag_load(aL1 + k0, half);
    v16h b0 = frag_load(b0p + k0, half);          // global (L2-resident)
    v16h b1 = frag_load(b1p + k0, half);
    v16h b2 = frag_load(b2p + k0, half);
    v16h b3 = frag_load(b3p + k0, half);
    acc[0][0] = __builtin_amdgcn_wmma_f32_16x16x32_f16(false, a0, false, b0, (short)0, acc[0][0], false, false);
    acc[0][1] = __builtin_amdgcn_wmma_f32_16x16x32_f16(false, a0, false, b1, (short)0, acc[0][1], false, false);
    acc[0][2] = __builtin_amdgcn_wmma_f32_16x16x32_f16(false, a0, false, b2, (short)0, acc[0][2], false, false);
    acc[0][3] = __builtin_amdgcn_wmma_f32_16x16x32_f16(false, a0, false, b3, (short)0, acc[0][3], false, false);
    acc[1][0] = __builtin_amdgcn_wmma_f32_16x16x32_f16(false, a1, false, b0, (short)0, acc[1][0], false, false);
    acc[1][1] = __builtin_amdgcn_wmma_f32_16x16x32_f16(false, a1, false, b1, (short)0, acc[1][1], false, false);
    acc[1][2] = __builtin_amdgcn_wmma_f32_16x16x32_f16(false, a1, false, b2, (short)0, acc[1][2], false, false);
    acc[1][3] = __builtin_amdgcn_wmma_f32_16x16x32_f16(false, a1, false, b3, (short)0, acc[1][3], false, false);
  }

#pragma unroll
  for (int mm = 0; mm < 2; ++mm) {
#pragma unroll
    for (int t = 0; t < 4; ++t) {
      int n = n0 + 16 * t + ln;
      float bv = bias[n];
#pragma unroll
      for (int r = 0; r < 8; ++r) {
        int m = m0 + mm * 16 + r + half * 8;   // C layout: VGPR r -> row r+8*half
        float val = acc[mm][t][r] + bv;
        if (mode == 3) {
          outF[(size_t)m * D_MODEL + n] = val;
        } else {
          int b_ = m >> 11;                    // m / SEQ
          int s_ = m & (SEQ - 1);
          int h_ = n >> 6;                     // n / HDIM
          int d_ = n & (HDIM - 1);
          int bh = b_ * NHEADS + h_;
          if (mode == 0)
            outH[((size_t)bh * SEQ + s_) * HDIM + d_] = (_Float16)(val * 0.125f);
          else if (mode == 1)
            outH[((size_t)bh * SEQ + s_) * HDIM + d_] = (_Float16)val;
          else
            outH[((size_t)bh * HDIM + d_) * SEQ + s_] = (_Float16)val;
        }
      }
    }
  }
}

// Flash attention: one wave per 16-query tile, kv tiles of 32 keys.
// Q pre-scaled by 1/sqrt(HDIM). Writes f16 output in (b, s, h*hd) layout.
__global__ __launch_bounds__(128)
void flash_attn(const _Float16* __restrict__ Q, const _Float16* __restrict__ K,
                const _Float16* __restrict__ Vt, _Float16* __restrict__ O) {
  __shared__ __align__(16) _Float16 Plds[4][16 * 32];   // per-wave 16x32 P staging
  const int lane = threadIdx.x & 31;
  const int wave = threadIdx.x >> 5;
  const int half = lane >> 4;
  const int ln   = lane & 15;
  const int bh   = blockIdx.y;
  const int q0   = (blockIdx.x * 4 + wave) * 16;

  const _Float16* Qb = Q + ((size_t)bh * SEQ + q0 + ln) * HDIM;
  v16h qa0 = frag_load(Qb, half);        // d = 0..31
  v16h qa1 = frag_load(Qb + 32, half);   // d = 32..63

  v8f   o[4] = {};
  float mi[8], li[8];
#pragma unroll
  for (int r = 0; r < 8; ++r) { mi[r] = -1e30f; li[r] = 0.0f; }

  for (int j0 = 0; j0 < q0 + 16; j0 += 32) {
    int klo = j0 + ln;
    int khi = j0 + 16 + ln;
    int kloC = klo < SEQ - 1 ? klo : SEQ - 1;   // clamp diag-tile overreads
    int khiC = khi < SEQ - 1 ? khi : SEQ - 1;

    const _Float16* Kl = K + ((size_t)bh * SEQ + kloC) * HDIM;
    const _Float16* Kh = K + ((size_t)bh * SEQ + khiC) * HDIM;

    // Batch all global loads (K frags + V frags) so they clause together and
    // overlap the softmax VALU work.
    v16h kb00 = frag_load(Kl, half);
    v16h kb01 = frag_load(Kl + 32, half);
    v16h kb10 = frag_load(Kh, half);
    v16h kb11 = frag_load(Kh + 32, half);
    v16h vb[4];
#pragma unroll
    for (int t = 0; t < 4; ++t)
      vb[t] = frag_load(Vt + ((size_t)bh * HDIM + 16 * t + ln) * SEQ + j0, half);

    v8f sl = {}, sh = {};
    sl = __builtin_amdgcn_wmma_f32_16x16x32_f16(false, qa0, false, kb00, (short)0, sl, false, false);
    sl = __builtin_amdgcn_wmma_f32_16x16x32_f16(false, qa1, false, kb01, (short)0, sl, false, false);
    sh = __builtin_amdgcn_wmma_f32_16x16x32_f16(false, qa0, false, kb10, (short)0, sh, false, false);
    sh = __builtin_amdgcn_wmma_f32_16x16x32_f16(false, qa1, false, kb11, (short)0, sh, false, false);

    float pl[8], ph[8], sc[8];
#pragma unroll
    for (int r = 0; r < 8; ++r) {
      int q = q0 + r + half * 8;
      float a = (klo <= q) ? sl[r] : -1e30f;    // causal mask: key > query
      float b = (khi <= q) ? sh[r] : -1e30f;
      float v = fmaxf(a, b);
#pragma unroll
      for (int msk = 1; msk < 16; msk <<= 1) v = fmaxf(v, __shfl_xor(v, msk));
      float mnew  = fmaxf(mi[r], v);
      float scale = __expf(mi[r] - mnew);
      float ea = __expf(a - mnew);
      float eb = __expf(b - mnew);
      float rs = ea + eb;
#pragma unroll
      for (int msk = 1; msk < 16; msk <<= 1) rs += __shfl_xor(rs, msk);
      li[r] = li[r] * scale + rs;
      mi[r] = mnew;
      sc[r] = scale;
      pl[r] = ea;
      ph[r] = eb;
    }

#pragma unroll
    for (int t = 0; t < 4; ++t)
#pragma unroll
      for (int r = 0; r < 8; ++r) o[t][r] *= sc[r];

    // C-layout -> A-layout relayout of P through LDS (same-wave, DS in order)
    asm volatile("" ::: "memory");
#pragma unroll
    for (int r = 0; r < 8; ++r) {
      int row = r + half * 8;
      Plds[wave][row * 32 + ln]      = (_Float16)pl[r];
      Plds[wave][row * 32 + 16 + ln] = (_Float16)ph[r];
    }
    asm volatile("s_wait_dscnt 0" ::: "memory");
    v16h pa = frag_load(&Plds[wave][ln * 32], half);
    asm volatile("" ::: "memory");

    // O += P x V : B[k][n] = V[key=j0+k][d], Vt is (bh, d, s) so keys contiguous
#pragma unroll
    for (int t = 0; t < 4; ++t)
      o[t] = __builtin_amdgcn_wmma_f32_16x16x32_f16(
          false, pa, false, vb[t], (short)0, o[t], false, false);
  }

  const int b_ = bh >> 4;
  const int h_ = bh & 15;
#pragma unroll
  for (int t = 0; t < 4; ++t)
#pragma unroll
    for (int r = 0; r < 8; ++r) {
      int s_ = q0 + r + half * 8;
      float val = o[t][r] / li[r];
      O[((size_t)(b_ * SEQ + s_)) * D_MODEL + h_ * HDIM + 16 * t + ln] = (_Float16)val;
    }
}

extern "C" void kernel_launch(void* const* d_in, const int* in_sizes, int n_in,
                              void* d_out, int out_size, void* d_ws, size_t ws_size,
                              hipStream_t stream) {
  const float* x  = (const float*)d_in[0];
  const float* Wq = (const float*)d_in[1];
  const float* bq = (const float*)d_in[2];
  const float* Wk = (const float*)d_in[3];
  const float* bk = (const float*)d_in[4];
  const float* Wv = (const float*)d_in[5];
  const float* bv = (const float*)d_in[6];
  const float* Wo = (const float*)d_in[7];
  const float* bo = (const float*)d_in[8];
  float* out = (float*)d_out;

  char* ws = (char*)d_ws;
  size_t off = 0;
  const size_t XSZ = (size_t)MTOT * D_MODEL * sizeof(_Float16);      // 16 MB
  const size_t WSZ = (size_t)D_MODEL * D_MODEL * sizeof(_Float16);   //  2 MB
  _Float16* xh  = (_Float16*)(ws + off); off += XSZ;
  _Float16* wqh = (_Float16*)(ws + off); off += WSZ;
  _Float16* wkh = (_Float16*)(ws + off); off += WSZ;
  _Float16* wvh = (_Float16*)(ws + off); off += WSZ;
  _Float16* woh = (_Float16*)(ws + off); off += WSZ;
  _Float16* qh  = (_Float16*)(ws + off); off += XSZ;
  _Float16* kh  = (_Float16*)(ws + off); off += XSZ;
  _Float16* vt  = (_Float16*)(ws + off); off += XSZ;
  _Float16* at  = (_Float16*)(ws + off); off += XSZ;   // attention output (b, s, h*hd)
  (void)in_sizes; (void)n_in; (void)out_size; (void)ws_size;

  f32_to_f16<<<4096, 256, 0, stream>>>(x,  xh,  MTOT * D_MODEL);
  f32_to_f16<<<1024, 256, 0, stream>>>(Wq, wqh, D_MODEL * D_MODEL);
  f32_to_f16<<<1024, 256, 0, stream>>>(Wk, wkh, D_MODEL * D_MODEL);
  f32_to_f16<<<1024, 256, 0, stream>>>(Wv, wvh, D_MODEL * D_MODEL);
  f32_to_f16<<<1024, 256, 0, stream>>>(Wo, woh, D_MODEL * D_MODEL);

  dim3 gg(MTOT / 32, D_MODEL / 256);
  gemm32<<<gg, 128, 0, stream>>>(xh, wqh, bq, qh, nullptr, 0);   // Q (scaled)
  gemm32<<<gg, 128, 0, stream>>>(xh, wkh, bk, kh, nullptr, 1);   // K
  gemm32<<<gg, 128, 0, stream>>>(xh, wvh, bv, vt, nullptr, 2);   // V^T

  flash_attn<<<dim3(SEQ / 64, BATCH * NHEADS), 128, 0, stream>>>(qh, kh, vt, at);

  gemm32<<<gg, 128, 0, stream>>>(at, woh, bo, nullptr, out, 3);  // out proj
}
